// GSSM_v5_35605278884494
// MI455X (gfx1250) — compile-verified, matrix operand order checked by reading, not run on hardware
//
#include <hip/hip_runtime.h>

// ---------------------------------------------------------------------------
// GSSM (voxel Mamba) for MI455X / gfx1250, wave32, WMMA bf16 GEMMs,
// chunked (associative) selective scan in f32.
// Round 3: fix async-to-LDS builtin pointee types (as1/as3 int4*).
// ---------------------------------------------------------------------------

#define D_MODEL 128
#define D_INNER 256
#define D_STATE 16
#define DT_RANK 8
#define BATCH   4
#define L_SEQ   16384
#define N_TOK   (BATCH * L_SEQ)
#define CHL     128                 // scan chunk length
#define NCH     (L_SEQ / CHL)       // 128 chunks per batch
#define LDC     132                 // padded f32 C-tile stride (bank-conflict free)

typedef __attribute__((ext_vector_type(16))) __bf16 v16bf;
typedef __attribute__((ext_vector_type(8)))  float  v8f;
typedef __attribute__((ext_vector_type(4)))  int    v4i;

#define DEV __device__ __forceinline__

#define WMMA_BF16(A_, B_, C_) \
  __builtin_amdgcn_wmma_f32_16x16x32_bf16(false, (A_), false, (B_), (short)0, (C_), false, false)

// ---- async global -> LDS staging (GLOBAL_LOAD_ASYNC_TO_LDS_B128) -----------
#if __has_builtin(__builtin_amdgcn_global_load_async_to_lds_b128)
#define HAVE_ASYNC_COPY 1
#else
#define HAVE_ASYNC_COPY 0
#endif

typedef __attribute__((address_space(1))) v4i g1_v4i;
typedef __attribute__((address_space(3))) v4i l3_v4i;

DEV void cp_b128(void* lds_dst, const void* gsrc) {
#if HAVE_ASYNC_COPY
  // generic shared ptr: low 32 bits are the LDS byte offset (aperture rules).
  __builtin_amdgcn_global_load_async_to_lds_b128(
      (g1_v4i*)(uintptr_t)gsrc,
      (l3_v4i*)(unsigned int)(uintptr_t)lds_dst, 0, 0);
#else
  *(uint4*)lds_dst = *(const uint4*)gsrc;
#endif
}

DEV void cp_fence() {
#if HAVE_ASYNC_COPY
#if __has_builtin(__builtin_amdgcn_s_wait_asynccnt)
  __builtin_amdgcn_s_wait_asynccnt(0);
#else
  asm volatile("s_wait_asynccnt 0" ::: "memory");
#endif
#endif
}

// ---- WMMA fragment helpers (wave32, 16x16x32 bf16) -------------------------
// A layout (16x32, MxK): lanes 0-15 hold M=lane, K in {0..7, 16..23};
// lanes 16-31 hold M=lane-16, K in {8..15, 24..31}.  B is loaded from a
// pre-transposed [N][K] matrix with the identical pattern (N plays M's role).
DEV v16bf frag_ab(const __bf16* tile, int ld) {
  const int lane = threadIdx.x & 31;
  const __bf16* p = tile + (lane & 15) * ld + ((lane >> 4) << 3);
  v16bf v;
#pragma unroll
  for (int e = 0; e < 8; ++e) { v[e] = p[e]; v[8 + e] = p[16 + e]; }
  return v;
}

// C/D layout: lane holds column n = lane&15; element r holds row m = r + 8*(lane>>4).
DEV void tile_to_lds(float* C, int ldc, int m0, int n0, v8f a) {
  const int lane = threadIdx.x & 31;
  const int n  = n0 + (lane & 15);
  const int mb = m0 + ((lane >> 4) << 3);
#pragma unroll
  for (int r = 0; r < 8; ++r) C[(mb + r) * ldc + n] = a[r];
}

DEV v8f tile_from_lds(const float* C, int ldc, int m0, int n0) {
  const int lane = threadIdx.x & 31;
  const int n  = n0 + (lane & 15);
  const int mb = m0 + ((lane >> 4) << 3);
  v8f a;
#pragma unroll
  for (int r = 0; r < 8; ++r) a[r] = C[(mb + r) * ldc + n];
  return a;
}

DEV void tile_to_global_bf16(__bf16* C, size_t ldc, size_t row0, int n0, v8f a) {
  const int lane = threadIdx.x & 31;
  const int n  = n0 + (lane & 15);
  const int mb = (lane >> 4) << 3;
#pragma unroll
  for (int r = 0; r < 8; ++r) C[(row0 + mb + r) * ldc + n] = (__bf16)a[r];
}

DEV float sigmoidf_(float x) { return 1.f / (1.f + __expf(-x)); }
DEV float softplusf_(float x) { return (x > 20.f) ? x : log1pf(__expf(x)); }

// ---------------------------------------------------------------------------
// Kernel 0: weight prep — bf16 transposes ([Nout][K] layout) + A = -exp(A_log)
// ---------------------------------------------------------------------------
__global__ __launch_bounds__(256) void k_prep(
    const float* conv_w, const float* pos_w2, const float* in_proj,
    const float* x_proj, const float* out_proj, const float* A_log,
    __bf16* convWT, __bf16* posW2T, __bf16* inProjT, __bf16* xProjT,
    __bf16* outProjT, float* Afac) {
  int i = blockIdx.x * 256 + threadIdx.x;
  if (i < 16384) { int o = i >> 7, k = i & 127; convWT[i]  = (__bf16)conv_w[k * 128 + o]; return; }
  i -= 16384;
  if (i < 16384) { int o = i >> 7, k = i & 127; posW2T[i]  = (__bf16)pos_w2[k * 128 + o]; return; }
  i -= 16384;
  if (i < 65536) { int o = i >> 7, k = i & 127; inProjT[i] = (__bf16)in_proj[k * 512 + o]; return; }
  i -= 65536;
  if (i < 12288) { int o = i >> 8, k = i & 255;
                   xProjT[i] = (__bf16)((o < 40) ? x_proj[k * 40 + o] : 0.f); return; }
  i -= 12288;
  if (i < 32768) { int o = i >> 8, k = i & 255; outProjT[i] = (__bf16)out_proj[k * 128 + o]; return; }
  i -= 32768;
  if (i < 4096)  { Afac[i] = -__expf(A_log[i]); }
}

// ---------------------------------------------------------------------------
// Kernel 1: fused front end.
//   h  = relu(LN(feats@conv_w + b))          (WMMA GEMM 1)
//   pe = relu(LN(pc@pos_w1 + b1))@pos_w2+b2  (VALU tiny layer + WMMA GEMM 2)
//   hn = rms(h + pe) * rms_g  -> scatter to Morton order (bf16)
// 128 rows / block, 8 waves; each wave owns a 16-row strip.
// ---------------------------------------------------------------------------
__global__ __launch_bounds__(256) void k_front(
    const float* feats, const int* coords, const int* inv_perm,
    const float* conv_b, const float* cln_g, const float* cln_b,
    const float* pos_w1, const float* pos_b1, const float* pln_g, const float* pln_b,
    const float* pos_b2, const float* rms_g,
    const __bf16* convWT, const __bf16* posW2T, __bf16* hn) {
  __shared__ __bf16 Ab[128 * 128];
  __shared__ __bf16 Wb[128 * 128];
  __shared__ float  Cl[128 * LDC];
  __shared__ float  redA[256], redB[256];

  const int tid  = threadIdx.x;
  const int wave = tid >> 5;
  const int n0   = blockIdx.x * 128;

  // stage feats tile (f32 -> bf16) and conv_w^T (async direct-to-LDS)
  for (int i = tid; i < 128 * 128; i += 256) {
    int r = i >> 7, c = i & 127;
    Ab[i] = (__bf16)feats[(size_t)(n0 + r) * 128 + c];
  }
  for (int i = tid; i < 2048; i += 256)
    cp_b128((void*)(Wb + i * 8), (const void*)(convWT + i * 8));
  cp_fence();
  __syncthreads();

  // GEMM 1: Cl = feats @ conv_w  (k outer, 8 independent WMMA chains inner)
  {
    v8f acc[8];
#pragma unroll
    for (int nt = 0; nt < 8; ++nt) acc[nt] = (v8f){};
#pragma unroll
    for (int k = 0; k < 4; ++k) {
      v16bf afr = frag_ab(&Ab[(wave * 16) * 128 + k * 32], 128);
      v16bf bfr[8];
#pragma unroll
      for (int nt = 0; nt < 8; ++nt) bfr[nt] = frag_ab(&Wb[(nt * 16) * 128 + k * 32], 128);
#pragma unroll
      for (int nt = 0; nt < 8; ++nt) acc[nt] = WMMA_BF16(afr, bfr[nt], acc[nt]);
    }
#pragma unroll
    for (int nt = 0; nt < 8; ++nt) tile_to_lds(Cl, LDC, wave * 16, nt * 16, acc[nt]);
  }
  __syncthreads();

  // bias + LayerNorm + relu on Cl rows (2 threads per row, 64 cols each)
  {
    const int r = tid >> 1, hf = tid & 1;
    float vals[64]; float s = 0.f, s2 = 0.f;
    for (int c = 0; c < 64; ++c) {
      int col = hf * 64 + c;
      float v = Cl[r * LDC + col] + conv_b[col];
      vals[c] = v; s += v; s2 += v * v;
    }
    redA[tid] = s; redB[tid] = s2;
    __syncthreads();
    float S  = redA[r * 2] + redA[r * 2 + 1];
    float S2 = redB[r * 2] + redB[r * 2 + 1];
    float m  = S * (1.f / 128.f);
    float rs = rsqrtf(S2 * (1.f / 128.f) - m * m + 1e-5f);
    __syncthreads();
    for (int c = 0; c < 64; ++c) {
      int col = hf * 64 + c;
      float v = (vals[c] - m) * rs * cln_g[col] + cln_b[col];
      Cl[r * LDC + col] = (v > 0.f) ? v : 0.f;
    }
  }
  __syncthreads();

  // positional-embedding hidden layer: relu(LN(pc@W1 + b1)) -> Ab (bf16)
  {
    const int r = tid >> 1, hf = tid & 1;
    const int n = n0 + r;
    float pz = (float)coords[n * 4 + 1];
    float py = (float)coords[n * 4 + 2];
    float px = (float)coords[n * 4 + 3];
    float vals[64]; float s = 0.f, s2 = 0.f;
    for (int c = 0; c < 64; ++c) {
      int col = hf * 64 + c;
      float v = pz * pos_w1[col] + py * pos_w1[128 + col] + px * pos_w1[256 + col] + pos_b1[col];
      vals[c] = v; s += v; s2 += v * v;
    }
    redA[tid] = s; redB[tid] = s2;
    __syncthreads();
    float S  = redA[r * 2] + redA[r * 2 + 1];
    float S2 = redB[r * 2] + redB[r * 2 + 1];
    float m  = S * (1.f / 128.f);
    float rs = rsqrtf(S2 * (1.f / 128.f) - m * m + 1e-5f);
    __syncthreads();
    for (int c = 0; c < 64; ++c) {
      int col = hf * 64 + c;
      float v = (vals[c] - m) * rs * pln_g[col] + pln_b[col];
      Ab[r * 128 + col] = (__bf16)((v > 0.f) ? v : 0.f);
    }
  }
  // restage W = pos_w2^T
  for (int i = tid; i < 2048; i += 256)
    cp_b128((void*)(Wb + i * 8), (const void*)(posW2T + i * 8));
  cp_fence();
  __syncthreads();

  // GEMM 2 (accumulate onto h): Cl += hidden @ pos_w2
  {
    v8f acc[8];
#pragma unroll
    for (int nt = 0; nt < 8; ++nt) acc[nt] = tile_from_lds(Cl, LDC, wave * 16, nt * 16);
#pragma unroll
    for (int k = 0; k < 4; ++k) {
      v16bf afr = frag_ab(&Ab[(wave * 16) * 128 + k * 32], 128);
      v16bf bfr[8];
#pragma unroll
      for (int nt = 0; nt < 8; ++nt) bfr[nt] = frag_ab(&Wb[(nt * 16) * 128 + k * 32], 128);
#pragma unroll
      for (int nt = 0; nt < 8; ++nt) acc[nt] = WMMA_BF16(afr, bfr[nt], acc[nt]);
    }
#pragma unroll
    for (int nt = 0; nt < 8; ++nt) tile_to_lds(Cl, LDC, wave * 16, nt * 16, acc[nt]);
  }
  __syncthreads();

  // f = h + pe (+pos_b2); RMSNorm; scatter to Morton (permuted) order
  {
    const int r = tid >> 1, hf = tid & 1;
    float vals[64]; float s2 = 0.f;
    for (int c = 0; c < 64; ++c) {
      int col = hf * 64 + c;
      float v = Cl[r * LDC + col] + pos_b2[col];
      vals[c] = v; s2 += v * v;
    }
    redB[tid] = s2;
    __syncthreads();
    float S2 = redB[r * 2] + redB[r * 2 + 1];
    float rs = rsqrtf(S2 * (1.f / 128.f) + 1e-5f);
    const int n = n0 + r;
    const int b = n >> 14;
    const size_t drow = (size_t)(b * L_SEQ + inv_perm[n]) * 128;
    for (int c = 0; c < 64; ++c) {
      int col = hf * 64 + c;
      hn[drow + col] = (__bf16)(vals[c] * rs * rms_g[col]);
    }
  }
}

// ---------------------------------------------------------------------------
// Kernel 2: xz = hn @ in_proj  (128x512 per 128-row tile), bf16 out.
// in_proj^T fully staged in LDS (128 KB) + A tile (32 KB) = 160 KB of 320 KB.
// 32 output tiles processed as 4 groups of 8 independent accumulator chains.
// ---------------------------------------------------------------------------
__global__ __launch_bounds__(256) void k_inproj(
    const __bf16* hn, const __bf16* inProjT, __bf16* xz) {
  __shared__ __bf16 Ab[128 * 128];
  __shared__ __bf16 Wb[512 * 128];
  const int tid  = threadIdx.x;
  const int wave = tid >> 5;
  const int n0   = blockIdx.x * 128;

  const __bf16* src = hn + (size_t)n0 * 128;
  for (int i = tid; i < 2048; i += 256) cp_b128((void*)(Ab + i * 8), (const void*)(src + i * 8));
  for (int i = tid; i < 8192; i += 256) cp_b128((void*)(Wb + i * 8), (const void*)(inProjT + i * 8));
  cp_fence();
  __syncthreads();

  for (int g = 0; g < 4; ++g) {
    v8f acc[8];
#pragma unroll
    for (int j = 0; j < 8; ++j) acc[j] = (v8f){};
#pragma unroll
    for (int k = 0; k < 4; ++k) {
      v16bf afr = frag_ab(&Ab[(wave * 16) * 128 + k * 32], 128);
      v16bf bfr[8];
#pragma unroll
      for (int j = 0; j < 8; ++j)
        bfr[j] = frag_ab(&Wb[((g * 8 + j) * 16) * 128 + k * 32], 128);
#pragma unroll
      for (int j = 0; j < 8; ++j) acc[j] = WMMA_BF16(afr, bfr[j], acc[j]);
    }
#pragma unroll
    for (int j = 0; j < 8; ++j)
      tile_to_global_bf16(xz, 512, (size_t)n0 + wave * 16, (g * 8 + j) * 16, acc[j]);
  }
}

// ---------------------------------------------------------------------------
// Kernel 3: causal depthwise conv1d (K=4) + bias + SiLU on the x half of xz.
// ---------------------------------------------------------------------------
__global__ __launch_bounds__(256) void k_dwconv(
    const __bf16* xz, const float* cw, const float* cb, __bf16* xcs) {
  const int idx = blockIdx.x * 256 + threadIdx.x;
  const int n = idx >> 8;
  const int d = idx & 255;
  const int l = n & (L_SEQ - 1);
  float acc = cb[d];
#pragma unroll
  for (int j = 0; j < 4; ++j) {
    int li = l - 3 + j;
    if (li >= 0) acc += cw[d * 4 + j] * (float)xz[(size_t)(n - 3 + j) * 512 + d];
  }
  xcs[(size_t)n * 256 + d] = (__bf16)(acc * sigmoidf_(acc));
}

// ---------------------------------------------------------------------------
// Kernel 4: xdbl = xcs @ x_proj (padded to 48 cols, WMMA), split B/C, and
// dt = softplus(dtp @ dt_w + dt_b) fused (tiny K=8 done on VALU).
// ---------------------------------------------------------------------------
#define LDX 52   // padded stride for the 48-col f32 tile
__global__ __launch_bounds__(256) void k_xproj(
    const __bf16* xcs, const __bf16* xProjT, const float* dt_w, const float* dt_b,
    float* dtb, float* Bm, float* Cm) {
  __shared__ __bf16 Ab[128 * 256];
  __shared__ __bf16 Wb[48 * 256];
  __shared__ float  Xd[128 * LDX];
  const int tid  = threadIdx.x;
  const int wave = tid >> 5;
  const int n0   = blockIdx.x * 128;

  const __bf16* src = xcs + (size_t)n0 * 256;
  for (int i = tid; i < 4096; i += 256) cp_b128((void*)(Ab + i * 8), (const void*)(src + i * 8));
  for (int i = tid; i < 1536; i += 256) cp_b128((void*)(Wb + i * 8), (const void*)(xProjT + i * 8));
  cp_fence();
  __syncthreads();

  {
    v8f acc[3];
#pragma unroll
    for (int nt = 0; nt < 3; ++nt) acc[nt] = (v8f){};
#pragma unroll
    for (int k = 0; k < 8; ++k) {
      v16bf afr = frag_ab(&Ab[(wave * 16) * 256 + k * 32], 256);
      v16bf bfr[3];
#pragma unroll
      for (int nt = 0; nt < 3; ++nt) bfr[nt] = frag_ab(&Wb[(nt * 16) * 256 + k * 32], 256);
#pragma unroll
      for (int nt = 0; nt < 3; ++nt) acc[nt] = WMMA_BF16(afr, bfr[nt], acc[nt]);
    }
#pragma unroll
    for (int nt = 0; nt < 3; ++nt) tile_to_lds(Xd, LDX, wave * 16, nt * 16, acc[nt]);
  }
  __syncthreads();

  // split out B / C state projections
  for (int i = tid; i < 128 * 32; i += 256) {
    int r = i >> 5, c = i & 31;
    float v = Xd[r * LDX + 8 + c];
    size_t n = (size_t)(n0 + r);
    if (c < 16) Bm[n * 16 + c] = v; else Cm[n * 16 + (c - 16)] = v;
  }
  // dt: thread d handles channel d across all 128 rows
  {
    const int d = tid;
    float wj[8];
#pragma unroll
    for (int j = 0; j < 8; ++j) wj[j] = dt_w[j * 256 + d];
    const float bb = dt_b[d];
    for (int r = 0; r < 128; ++r) {
      float s = bb;
#pragma unroll
      for (int j = 0; j < 8; ++j) s += Xd[r * LDX + j] * wj[j];
      dtb[(size_t)(n0 + r) * 256 + d] = softplusf_(s);
    }
  }
}

// ---------------------------------------------------------------------------
// Chunked selective scan.  Recurrence h = exp(dt*A)*h + (dt*x)*B is linear and
// associative; the chunk decay collapses to exp(A * sum(dt)).
// ---------------------------------------------------------------------------
__global__ __launch_bounds__(256) void k_scan_chunk(
    const float* dtb, const __bf16* xcs, const float* Bm, const float* Afac,
    float* hend, float* dtsum) {
  const int blk = blockIdx.x;
  const int b = blk >> 7;
  const int c = blk & (NCH - 1);
  const int d = threadIdx.x;
  float a[16], h[16];
#pragma unroll
  for (int s = 0; s < 16; ++s) { a[s] = Afac[d * 16 + s]; h[s] = 0.f; }
  float dsum = 0.f;
  const size_t nbase = (size_t)b * L_SEQ + (size_t)c * CHL;
  for (int i = 0; i < CHL; ++i) {
    const size_t n = nbase + i;
    if (i + 8 < CHL) {
      __builtin_prefetch(dtb + (n + 8) * 256 + d, 0, 1);
      __builtin_prefetch(xcs + (n + 8) * 256 + d, 0, 1);
    }
    const float dtv = dtb[n * 256 + d];
    const float xv  = (float)xcs[n * 256 + d];
    const float dx  = dtv * xv;
    dsum += dtv;
    const float* Bp = Bm + n * 16;
#pragma unroll
    for (int s = 0; s < 16; ++s) h[s] = h[s] * __expf(a[s] * dtv) + dx * Bp[s];
  }
  const size_t base = (size_t)blk * 256 + d;
#pragma unroll
  for (int s = 0; s < 16; ++s) hend[base * 16 + s] = h[s];
  dtsum[base] = dsum;
}

__global__ __launch_bounds__(256) void k_scan_prefix(
    const float* Afac, const float* hend, const float* dtsum, float* hin) {
  const int b = blockIdx.x;
  const int d = threadIdx.x;
  float a[16], carry[16];
#pragma unroll
  for (int s = 0; s < 16; ++s) { a[s] = Afac[d * 16 + s]; carry[s] = 0.f; }
  for (int c = 0; c < NCH; ++c) {
    const size_t base = (size_t)(b * NCH + c) * 256 + d;
#pragma unroll
    for (int s = 0; s < 16; ++s) hin[base * 16 + s] = carry[s];
    const float dsv = dtsum[base];
#pragma unroll
    for (int s = 0; s < 16; ++s)
      carry[s] = carry[s] * __expf(a[s] * dsv) + hend[base * 16 + s];
  }
}

__global__ __launch_bounds__(256) void k_scan_final(
    const float* dtb, const __bf16* xcs, const float* Bm, const float* Cm,
    const float* Afac, const float* hin, const __bf16* xz, const float* Dp,
    __bf16* yg) {
  const int blk = blockIdx.x;
  const int b = blk >> 7;
  const int c = blk & (NCH - 1);
  const int d = threadIdx.x;
  float a[16], h[16];
#pragma unroll
  for (int s = 0; s < 16; ++s) a[s] = Afac[d * 16 + s];
  const size_t sbase = ((size_t)blk * 256 + d) * 16;
#pragma unroll
  for (int s = 0; s < 16; ++s) h[s] = hin[sbase + s];
  const float dp = Dp[d];
  const size_t nbase = (size_t)b * L_SEQ + (size_t)c * CHL;
  for (int i = 0; i < CHL; ++i) {
    const size_t n = nbase + i;
    if (i + 8 < CHL) {
      __builtin_prefetch(dtb + (n + 8) * 256 + d, 0, 1);
      __builtin_prefetch(xcs + (n + 8) * 256 + d, 0, 1);
    }
    const float dtv = dtb[n * 256 + d];
    const float xv  = (float)xcs[n * 256 + d];
    const float dx  = dtv * xv;
    const float* Bp = Bm + n * 16;
    const float* Cp = Cm + n * 16;
    float y = 0.f;
#pragma unroll
    for (int s = 0; s < 16; ++s) {
      h[s] = h[s] * __expf(a[s] * dtv) + dx * Bp[s];
      y += h[s] * Cp[s];
    }
    y += xv * dp;
    const float zv = (float)xz[n * 512 + 256 + d];
    y *= zv * sigmoidf_(zv);
    yg[n * 256 + d] = (__bf16)y;
  }
}

// ---------------------------------------------------------------------------
// Kernel 6: out = LN( gather(yg, inv_perm) @ out_proj ).  K=256 split in two
// 128-wide phases so LDS stays bounded; WMMA accs persist in registers.
// ---------------------------------------------------------------------------
__global__ __launch_bounds__(256) void k_outproj(
    const __bf16* yg, const __bf16* outProjT, const int* inv_perm,
    const float* oln_g, const float* oln_b, float* out) {
  __shared__ __bf16 Ab[128 * 128];
  __shared__ __bf16 Wb[128 * 128];
  __shared__ float  Cl[128 * LDC];
  __shared__ int    srcRow[128];
  __shared__ float  redA[256], redB[256];
  const int tid  = threadIdx.x;
  const int wave = tid >> 5;
  const int n0   = blockIdx.x * 128;

  if (tid < 128) {
    const int n = n0 + tid;
    const int b = n >> 14;
    srcRow[tid] = b * L_SEQ + inv_perm[n];
  }

  v8f acc[8];
#pragma unroll
  for (int nt = 0; nt < 8; ++nt) acc[nt] = (v8f){};

  for (int ph = 0; ph < 2; ++ph) {
    __syncthreads();
    // gather A rows (16B chunks) + weight half, async direct to LDS
    for (int i = tid; i < 2048; i += 256) {
      int r = i >> 4, cc = (i & 15) * 8;
      cp_b128((void*)(Ab + r * 128 + cc),
              (const void*)(yg + (size_t)srcRow[r] * 256 + ph * 128 + cc));
    }
    for (int i = tid; i < 2048; i += 256) {
      int o = i >> 4, cc = (i & 15) * 8;
      cp_b128((void*)(Wb + o * 128 + cc),
              (const void*)(outProjT + o * 256 + ph * 128 + cc));
    }
    cp_fence();
    __syncthreads();
#pragma unroll
    for (int k = 0; k < 4; ++k) {
      v16bf afr = frag_ab(&Ab[(wave * 16) * 128 + k * 32], 128);
      v16bf bfr[8];
#pragma unroll
      for (int nt = 0; nt < 8; ++nt) bfr[nt] = frag_ab(&Wb[(nt * 16) * 128 + k * 32], 128);
#pragma unroll
      for (int nt = 0; nt < 8; ++nt) acc[nt] = WMMA_BF16(afr, bfr[nt], acc[nt]);
    }
  }
  __syncthreads();
#pragma unroll
  for (int nt = 0; nt < 8; ++nt) tile_to_lds(Cl, LDC, wave * 16, nt * 16, acc[nt]);
  __syncthreads();

  // final LayerNorm, f32 output
  {
    const int r = tid >> 1, hf = tid & 1;
    float vals[64]; float s = 0.f, s2 = 0.f;
    for (int c = 0; c < 64; ++c) {
      int col = hf * 64 + c;
      float v = Cl[r * LDC + col];
      vals[c] = v; s += v; s2 += v * v;
    }
    redA[tid] = s; redB[tid] = s2;
    __syncthreads();
    float S  = redA[r * 2] + redA[r * 2 + 1];
    float S2 = redB[r * 2] + redB[r * 2 + 1];
    float m  = S * (1.f / 128.f);
    float rs = rsqrtf(S2 * (1.f / 128.f) - m * m + 1e-5f);
    for (int c = 0; c < 64; ++c) {
      int col = hf * 64 + c;
      out[(size_t)(n0 + r) * 128 + col] = (vals[c] - m) * rs * oln_g[col] + oln_b[col];
    }
  }
}

// ---------------------------------------------------------------------------
extern "C" void kernel_launch(void* const* d_in, const int* in_sizes, int n_in,
                              void* d_out, int out_size, void* d_ws, size_t ws_size,
                              hipStream_t stream) {
  (void)in_sizes; (void)n_in; (void)out_size; (void)ws_size;
  const float* feats    = (const float*)d_in[0];
  const int*   coords   = (const int*)d_in[1];
  const int*   inv_perm = (const int*)d_in[3];
  const float* conv_w   = (const float*)d_in[4];
  const float* conv_b   = (const float*)d_in[5];
  const float* cln_g    = (const float*)d_in[6];
  const float* cln_b    = (const float*)d_in[7];
  const float* pos_w1   = (const float*)d_in[8];
  const float* pos_b1   = (const float*)d_in[9];
  const float* pln_g    = (const float*)d_in[10];
  const float* pln_b    = (const float*)d_in[11];
  const float* pos_w2   = (const float*)d_in[12];
  const float* pos_b2   = (const float*)d_in[13];
  const float* rms_g    = (const float*)d_in[14];
  const float* in_proj  = (const float*)d_in[15];
  const float* conv1d_w = (const float*)d_in[16];
  const float* conv1d_b = (const float*)d_in[17];
  const float* x_proj   = (const float*)d_in[18];
  const float* dt_w     = (const float*)d_in[19];
  const float* dt_b     = (const float*)d_in[20];
  const float* A_log    = (const float*)d_in[21];
  const float* Dp       = (const float*)d_in[22];
  const float* out_proj = (const float*)d_in[23];
  const float* oln_g    = (const float*)d_in[24];
  const float* oln_b    = (const float*)d_in[25];
  float* out = (float*)d_out;

  char* ws = (char*)d_ws;
  size_t off = 0;
  auto alloc = [&](size_t bytes) { size_t r = off; off = (off + bytes + 255) & ~(size_t)255; return r; };

  __bf16* convWT   = (__bf16*)(ws + alloc(128 * 128 * 2));
  __bf16* posW2T   = (__bf16*)(ws + alloc(128 * 128 * 2));
  __bf16* inProjT  = (__bf16*)(ws + alloc(512 * 128 * 2));
  __bf16* xProjT   = (__bf16*)(ws + alloc(48 * 256 * 2));
  __bf16* outProjT = (__bf16*)(ws + alloc(128 * 256 * 2));
  float*  Afac     = (float*)(ws + alloc(256 * 16 * 4));
  __bf16* hn       = (__bf16*)(ws + alloc((size_t)N_TOK * 128 * 2));
  __bf16* xz       = (__bf16*)(ws + alloc((size_t)N_TOK * 512 * 2));
  __bf16* xcs      = (__bf16*)(ws + alloc((size_t)N_TOK * 256 * 2));
  float*  dtb      = (float*)(ws + alloc((size_t)N_TOK * 256 * 4));
  float*  Bm       = (float*)(ws + alloc((size_t)N_TOK * 16 * 4));
  float*  Cm       = (float*)(ws + alloc((size_t)N_TOK * 16 * 4));
  float*  hend     = (float*)(ws + alloc((size_t)BATCH * NCH * 256 * 16 * 4));
  float*  dtsum    = (float*)(ws + alloc((size_t)BATCH * NCH * 256 * 4));
  float*  hin      = (float*)(ws + alloc((size_t)BATCH * NCH * 256 * 16 * 4));
  __bf16* yg       = (__bf16*)(ws + alloc((size_t)N_TOK * 256 * 2));

  k_prep<<<576, 256, 0, stream>>>(conv_w, pos_w2, in_proj, x_proj, out_proj, A_log,
                                  convWT, posW2T, inProjT, xProjT, outProjT, Afac);
  k_front<<<N_TOK / 128, 256, 0, stream>>>(feats, coords, inv_perm,
                                           conv_b, cln_g, cln_b,
                                           pos_w1, pos_b1, pln_g, pln_b,
                                           pos_b2, rms_g, convWT, posW2T, hn);
  k_inproj<<<N_TOK / 128, 256, 0, stream>>>(hn, inProjT, xz);
  k_dwconv<<<(size_t)N_TOK * 256 / 256, 256, 0, stream>>>(xz, conv1d_w, conv1d_b, xcs);
  k_xproj<<<N_TOK / 128, 256, 0, stream>>>(xcs, xProjT, dt_w, dt_b, dtb, Bm, Cm);
  k_scan_chunk<<<BATCH * NCH, 256, 0, stream>>>(dtb, xcs, Bm, Afac, hend, dtsum);
  k_scan_prefix<<<BATCH, 256, 0, stream>>>(Afac, hend, dtsum, hin);
  k_scan_final<<<BATCH * NCH, 256, 0, stream>>>(dtb, xcs, Bm, Cm, Afac, hin, xz, Dp, yg);
  k_outproj<<<N_TOK / 128, 256, 0, stream>>>(yg, outProjT, inv_perm, oln_g, oln_b, out);
}
